// HyperGC_74990128988679
// MI455X (gfx1250) — matrix operand order repaired and checked by reading, not compiled, not run on hardware
//
#include <hip/hip_runtime.h>
#include <hip/hip_bf16.h>
#include <math.h>

typedef __attribute__((ext_vector_type(16))) _Float16 v16h;
typedef __attribute__((ext_vector_type(16), aligned(16))) _Float16 v16hA; // 16B-aligned view
typedef __attribute__((ext_vector_type(2)))  _Float16 v2h;
typedef __attribute__((ext_vector_type(8)))  float    v8f;

constexpr int N_  = 64;
constexpr int C_  = 256;
constexpr int T_  = 64;
constexpr int V_  = 25;
constexpr int K_  = 8;
constexpr int VN_ = 5;
constexpr int VT_ = 30;
constexpr int CG_ = 32;   // C_/K_
constexpr int H_  = 8;    // CG_/4
constexpr int O_  = 256;
constexpr int OG_ = 32;   // O_/K_
constexpr int TOPK_ = 9;
constexpr int TTILE = 16; // t-tile in fused kernel

// ---- workspace layout (float offsets) ----
constexpr size_t WS_POOLED = 0;                                  // N,K,CG,VT f32
constexpr size_t POOLED_SZ = (size_t)N_*K_*CG_*VT_;
constexpr size_t WS_INCID  = WS_POOLED + POOLED_SZ;              // N,K,VT,VT f32
constexpr size_t INCID_SZ  = (size_t)N_*K_*VT_*VT_;
constexpr size_t WS_W1     = WS_INCID + INCID_SZ;                // N,K,H,VT f32
constexpr size_t W1_SZ     = (size_t)N_*K_*H_*VT_;
constexpr size_t WS_ADJF   = WS_W1 + W1_SZ;                      // N,K,32,32 f16 (zero padded)
constexpr size_t ADJF_SZF  = (size_t)N_*K_*32*32/2;              // in floats
constexpr size_t WS_WF16   = WS_ADJF + ADJF_SZF;                 // K,OG,CG f16

// -------------------------------------------------------------------------
// Kernel A: pooled[n,k,cg,v] = mean_t xv[n,c,t,v]   (v>=25 -> hyper_joint)
// -------------------------------------------------------------------------
__global__ void pool_kernel(const float* __restrict__ x,
                            const float* __restrict__ hyper_joint,
                            float* __restrict__ pooled) {
  int b = blockIdx.x;              // n*C_ + c
  int n = b / C_, c = b % C_;
  int lane = threadIdx.x;
  float s = 0.f;
  if (lane < V_) {
    const float* xp = x + ((size_t)(n*C_ + c))*T_*V_ + lane;
    #pragma unroll 4
    for (int t = 0; t < T_; ++t) s += xp[(size_t)t*V_];
    s *= (1.0f / T_);
  } else if (lane < VT_) {
    s = hyper_joint[(lane - V_)*C_ + c];
  }
  if (lane < VT_) {
    int k = c / CG_, cg = c % CG_;
    pooled[(((size_t)(n*K_ + k))*CG_ + cg)*VT_ + lane] = s;
  }
}

// prep: tuple-output tail copy + conv weight f32->f16 (enables async A-operand)
__global__ void prep_kernel(const float* __restrict__ hj,
                            const float* __restrict__ conv_d_w,
                            float* __restrict__ out_tail,
                            _Float16* __restrict__ wf16) {
  int i = blockIdx.x*blockDim.x + threadIdx.x;
  if (i < K_*OG_*CG_) wf16[i] = (_Float16)conv_d_w[i];
  int j = i - K_*OG_*CG_;
  if (j >= 0 && j < VN_*C_) out_tail[j] = hj[j];
}

// -------------------------------------------------------------------------
// Kernel B1: per (n,k): vproj -> dist -> top9 softmax incidence; and w1
// -------------------------------------------------------------------------
__global__ void hyper_b1_kernel(const float* __restrict__ pooled,
                                const float* __restrict__ to_v_w,
                                const float* __restrict__ to_v_b,
                                const float* __restrict__ to_w1_w,
                                const float* __restrict__ to_w1_b,
                                float* __restrict__ incid,
                                float* __restrict__ w1out) {
  int b = blockIdx.x; int n = b / K_, k = b % K_;
  int tid = threadIdx.x;
  __shared__ float pg[CG_][VT_];
  __shared__ float vp[VT_][H_];
  __shared__ float ds[VT_][VT_];

  const float* pgg = pooled + ((size_t)(n*K_ + k))*CG_*VT_;
  for (int i = tid; i < CG_*VT_; i += blockDim.x) pg[i/VT_][i%VT_] = pgg[i];
  __syncthreads();

  for (int i = tid; i < VT_*H_; i += blockDim.x) {
    int v = i / H_, h = i % H_;
    float s = to_v_b[k*H_ + h];
    const float* wr = to_v_w + (size_t)(k*H_ + h)*CG_;
    for (int c = 0; c < CG_; ++c) s += pg[c][v]*wr[c];
    vp[v][h] = s;
  }
  __syncthreads();

  for (int i = tid; i < VT_*VT_; i += blockDim.x) {
    int u = i / VT_, v = i % VT_;
    float d2 = 0.f;
    #pragma unroll
    for (int h = 0; h < H_; ++h) { float df = vp[u][h]-vp[v][h]; d2 += df*df; }
    ds[u][v] = (d2 > 0.f) ? sqrtf(d2) : 0.f;
  }
  __syncthreads();

  // top-9 smallest dist per row (ties -> lowest index), softmax of negatives
  if (tid < VT_) {
    int u = tid;
    unsigned mask = 0;
    int selv[TOPK_]; float vals[TOPK_];
    for (int j = 0; j < TOPK_; ++j) {
      float best = 3.4e38f; int bi = 0;
      for (int v = 0; v < VT_; ++v)
        if (!((mask >> v) & 1u) && ds[u][v] < best) { best = ds[u][v]; bi = v; }
      mask |= (1u << bi); selv[j] = bi; vals[j] = -best;
    }
    float m = vals[0];
    #pragma unroll
    for (int j = 1; j < TOPK_; ++j) m = fmaxf(m, vals[j]);
    float ex[TOPK_]; float ssum = 0.f;
    #pragma unroll
    for (int j = 0; j < TOPK_; ++j) { ex[j] = expf(vals[j]-m); ssum += ex[j]; }
    float inv = 1.f/ssum;
    float* ip = incid + (((size_t)(n*K_ + k))*VT_ + u)*VT_;
    for (int v = 0; v < VT_; ++v) ip[v] = 0.f;
    #pragma unroll
    for (int j = 0; j < TOPK_; ++j) ip[selv[j]] = ex[j]*inv;
  }

  for (int i = tid; i < H_*VT_; i += blockDim.x) {
    int h = i / VT_, v = i % VT_;
    float s = to_w1_b[k*H_ + h];
    const float* wr = to_w1_w + (size_t)(k*H_ + h)*CG_;
    for (int c = 0; c < CG_; ++c) s += pg[c][v]*wr[c];
    s = (s > 0.f) ? s : 0.01f*s;
    w1out[(((size_t)(n*K_ + k))*H_ + h)*VT_ + v] = s;
  }
}

// -------------------------------------------------------------------------
// Kernel B2: weights (cross-k) -> d, inc_w, hyper_adj -> adj_full (f16 pad32)
// -------------------------------------------------------------------------
__global__ void hyper_b2_kernel(const float* __restrict__ incid,
                                const float* __restrict__ w1,
                                const float* __restrict__ to_w2_w,
                                const float* __restrict__ to_w2_b,
                                const float* __restrict__ adjacency,
                                const float* __restrict__ edge_importance,
                                const float* __restrict__ alpha,
                                _Float16* __restrict__ adjf) {
  int b = blockIdx.x; int n = b / K_, k = b % K_;
  int tid = threadIdx.x;
  __shared__ float inc[VT_][VT_];
  __shared__ float w[VT_], dd[VT_], rsu[VT_], rsadj[VT_];

  const float* ip = incid + ((size_t)(n*K_ + k))*VT_*VT_;
  for (int i = tid; i < VT_*VT_; i += 32) inc[i/VT_][i%VT_] = ip[i];
  __syncthreads();

  if (tid < VT_) {
    int v = tid;
    float s = to_w2_b[k];
    for (int q = 0; q < K_*H_; ++q) {
      int kp = q / H_, h = q % H_;
      s += w1[(((size_t)(n*K_ + kp))*H_ + h)*VT_ + v] * to_w2_w[k*K_*H_ + q];
    }
    w[v] = tanhf(s);
  }
  __syncthreads();

  if (tid < VT_) {
    int v = tid;
    float ca = 0.f;
    for (int u = 0; u < VT_; ++u) ca += fabsf(inc[u][v]);
    dd[v] = w[v] / (ca + 1e-8f);
  }
  if (tid < VT_) {
    int u = tid;
    float rs = 0.f;
    for (int v = 0; v < VT_; ++v) rs += fabsf(inc[u][v]*w[v]);
    rsu[u] = rs + 1e-8f;
    const float* ar = adjacency + ((size_t)k*VT_ + u)*VT_;
    const float* er = edge_importance + ((size_t)k*VT_ + u)*VT_;
    float s = 0.f;
    for (int wc = 0; wc < VT_; ++wc) s += fabsf(ar[wc]*er[wc]);
    rsadj[u] = s + 1e-8f;
  }
  __syncthreads();

  float al = fmaxf(alpha[0], 0.f);
  _Float16* op = adjf + ((size_t)(n*K_ + k))*1024;
  for (int i = tid; i < 32*32; i += 32) {
    int u = i >> 5, wc = i & 31;
    float af = 0.f;
    if (u < VT_ && wc < VT_) {
      float ha = 0.f;
      float rinv = 1.f / rsu[u];
      for (int v = 0; v < VT_; ++v)
        ha += (inc[u][v]*w[v]*rinv) * dd[v] * inc[wc][v];
      size_t ai = ((size_t)k*VT_ + u)*VT_ + wc;
      float a0 = adjacency[ai]*edge_importance[ai] / rsadj[u];
      af = a0 + al*ha;
    }
    op[i] = (_Float16)af;
  }
}

// -------------------------------------------------------------------------
// Kernel C: fused dual-GEMM via v_wmma_f32_16x16x32_f16 + BN/residual/ReLU
// grid = N*K*(T/TTILE), block = 256 (8 waves)
// A-operands DMA'd global->LDS with global_load_async_to_lds_b128 (ASYNCcnt);
// B fragments contiguous 32B per lane via transposed LDS tiles.
// -------------------------------------------------------------------------
__global__ __launch_bounds__(256) void fused_gemm_kernel(
    const float* __restrict__ x,
    const float* __restrict__ hyper_joint,
    const _Float16* __restrict__ wf16,
    const float* __restrict__ conv_d_b,
    const float* __restrict__ bn_gamma,
    const float* __restrict__ bn_beta,
    const _Float16* __restrict__ adjf,
    float* __restrict__ out) {
  constexpr int NTCH = T_ / TTILE;           // 4
  constexpr int XROW = 40;                   // padded row stride (halves): 80B, 16B-aligned, bank-conflict-free
  int b  = blockIdx.x;
  int tc = b % NTCH;
  int k  = (b / NTCH) % K_;
  int n  = b / (NTCH * K_);
  int t0 = tc * TTILE;
  int tid  = threadIdx.x;
  int wave = tid >> 5, lane = tid & 31;
  int hf   = lane >> 4, l16 = lane & 15;     // half-wave, lane-in-half

  // XhT[tv][c]: X tile transposed; DhT[ot][v]: dense transposed (stage-2 B)
  struct S1 { _Float16 XhT[512][XROW]; _Float16 DhT[512][XROW]; };
  __shared__ union alignas(16) SU { S1 s1; float Ys[32][OG_*TTILE + 1]; } sm;
  __shared__ alignas(16) _Float16 Wh[OG_][CG_];
  __shared__ alignas(16) _Float16 A2h[32][32];
  __shared__ float bias_s[OG_];
  __shared__ float scale_s[OG_], beta_s[OG_];

  // ---- async DMA of the two A-operands straight into LDS (no VGPR round trip)
  if (wave < 2) {
    const _Float16* src = (wave == 0) ? (adjf + ((size_t)(n*K_ + k))*1024)
                                      : (wf16 + (size_t)k*OG_*CG_);
    unsigned ldsbase = (unsigned)(size_t)((wave == 0) ? &A2h[0][0] : &Wh[0][0]);
    #pragma unroll
    for (int it = 0; it < 4; ++it) {
      unsigned off = it*512u + (unsigned)lane*16u;     // 4 x 512B = 2KB
      asm volatile("global_load_async_to_lds_b128 %0, %1, %2"
                   :: "v"(ldsbase + off), "v"(off), "s"(src) : "memory");
    }
    asm volatile("s_wait_asynccnt 0" ::: "memory");    // done before barrier
  }

  // ---- stage 0: load X tile into LDS (f16 convert), transposed ----
  for (int i = tid; i < 512*(CG_/2); i += 256) {
    int tv = i & 511;                 // tl*32 + v
    int cp = i >> 9;                  // 0..15
    int c0 = cp << 1;
    int tl = tv >> 5, v = tv & 31;
    int ch0 = k*CG_ + c0;
    float f0 = 0.f, f1 = 0.f;
    if (v < V_) {
      size_t base = (((size_t)(n*C_ + ch0))*T_ + (t0+tl))*V_ + v;
      f0 = x[base];
      f1 = x[base + (size_t)T_*V_];
    } else if (v < VT_) {
      f0 = hyper_joint[(v - V_)*C_ + ch0];
      f1 = hyper_joint[(v - V_)*C_ + ch0 + 1];
    }
    v2h p; p[0] = (_Float16)f0; p[1] = (_Float16)f1;
    *(v2h*)(&sm.s1.XhT[tv][c0]) = p;
  }
  if (tid < OG_) {
    bias_s[tid]  = conv_d_b[k*OG_ + tid];
    scale_s[tid] = bn_gamma[k*OG_ + tid] * rsqrtf(1.0f + 1e-5f);
    beta_s[tid]  = bn_beta[k*OG_ + tid];
  }
  __syncthreads();

  int mt = wave >> 2;          // waves 0-3: m-tile 0; waves 4-7: m-tile 1 (wave-uniform)
  int m0 = mt * 16;

  // A fragment for stage 1 (conv weight), hoisted: wave-uniform per m-tile
  v16h aW;
  #pragma unroll
  for (int j = 0; j < 16; ++j) {
    int r = j >> 1, e = j & 1;
    int kk = hf*16 + j;    // contiguous k order, matches B
    aW[j] = Wh[m0 + l16][kk];
  }

  // ---- stage 1: dense[o][tl*32+v] = sum_c W[o][c]*X[c][tl*32+v]  (K=32) ----
  // software-pipelined: store of tile i-1 overlaps WMMA of tile i (hides
  // the 4-coexec WMMA->VALU hazard instead of v_nops).
  // X is zero at v>=30 so acc==0 there; A2h cols 30/31 are zero, so the
  // stray bias at padded v contributes exactly 0 in stage 2. No predication.
  v8f accP;
  int colP = 0;
  #pragma unroll
  for (int i = 0; i < 9; ++i) {
    v8f acc;
    if (i < 8) {
      int n0 = ((wave & 3)*8 + i)*16;
      v16h bb = *(const v16hA*)(&sm.s1.XhT[n0 + l16][hf*16]);   // contiguous 32B
      v8f cz = {};
      acc = __builtin_amdgcn_wmma_f32_16x16x32_f16(false, aW, false, bb,
                                                   (short)0, cz, false, false);
    }
    if (i > 0) {
      int tl = colP >> 5, v = colP & 31;
      #pragma unroll
      for (int r = 0; r < 8; ++r) {
        int o = m0 + r + hf*8;                                  // C/D: M = r + 8*hf
        sm.s1.DhT[o*TTILE + tl][v] = (_Float16)(accP[r] + bias_s[o]);
      }
    }
    if (i < 8) { accP = acc; colP = ((wave & 3)*8 + i)*16 + l16; }
  }
  __syncthreads();

  // A fragment for stage 2 (adj_full)
  v16h aA;
  #pragma unroll
  for (int j = 0; j < 16; ++j) {
    int r = j >> 1, e = j & 1;
    int kk = hf*16 + j;
    aA[j] = A2h[m0 + l16][kk];
  }

  // ---- stage 2: Y[u][ot] = sum_v adj_full[u][v]*dense_T[ot][v]  (K=32) ----
  v8f acc2[8];
  #pragma unroll
  for (int i = 0; i < 8; ++i) {
    int n0 = ((wave & 3)*8 + i)*16;
    v16h bb = *(const v16hA*)(&sm.s1.DhT[n0 + l16][hf*16]);     // contiguous 32B
    v8f cz = {};
    acc2[i] = __builtin_amdgcn_wmma_f32_16x16x32_f16(false, aA, false, bb,
                                                     (short)0, cz, false, false);
  }
  __syncthreads();   // all waves done reading XhT/DhT before Ys overlay

  #pragma unroll
  for (int i = 0; i < 8; ++i) {
    int ot = ((wave & 3)*8 + i)*16 + l16;
    #pragma unroll
    for (int r = 0; r < 8; ++r) {
      int u = m0 + r + hf*8;
      sm.Ys[u][ot] = acc2[i][r];
    }
  }
  __syncthreads();

  // ---- epilogue: BN scale/shift + residual + ReLU, coalesced in u ----
  for (int i = tid; i < OG_*TTILE*V_; i += 256) {
    int u   = i % V_;
    int tmp = i / V_;
    int tl  = tmp % TTILE;
    int o   = tmp / TTILE;
    float y = sm.Ys[u][o*TTILE + tl] * scale_s[o] + beta_s[o];
    size_t gi = (((size_t)(n*C_ + k*OG_ + o))*T_ + (t0+tl))*V_ + u;
    out[gi] = fmaxf(y + x[gi], 0.f);
  }
}

// -------------------------------------------------------------------------
extern "C" void kernel_launch(void* const* d_in, const int* in_sizes, int n_in,
                              void* d_out, int out_size, void* d_ws, size_t ws_size,
                              hipStream_t stream) {
  (void)in_sizes; (void)n_in; (void)out_size; (void)ws_size;
  const float* x               = (const float*)d_in[0];
  const float* adjacency       = (const float*)d_in[1];
  const float* edge_importance = (const float*)d_in[2];
  const float* hyper_joint     = (const float*)d_in[3];
  const float* alpha           = (const float*)d_in[4];
  const float* to_v_w          = (const float*)d_in[5];
  const float* to_v_b          = (const float*)d_in[6];
  const float* to_w1_w         = (const float*)d_in[7];
  const float* to_w1_b         = (const float*)d_in[8];
  const float* to_w2_w         = (const float*)d_in[9];
  const float* to_w2_b         = (const float*)d_in[10];
  const float* conv_d_w        = (const float*)d_in[11];
  const float* conv_d_b        = (const float*)d_in[12];
  const float* bn_gamma        = (const float*)d_in[13];
  const float* bn_beta         = (const float*)d_in[14];
  float* out = (float*)d_out;
  float* ws  = (float*)d_ws;

  float*    pooled = ws + WS_POOLED;
  float*    incid  = ws + WS_INCID;
  float*    w1     = ws + WS_W1;
  _Float16* adjf   = (_Float16*)(ws + WS_ADJF);
  _Float16* wf16   = (_Float16*)(ws + WS_WF16);

  pool_kernel<<<N_*C_, 32, 0, stream>>>(x, hyper_joint, pooled);
  prep_kernel<<<(K_*OG_*CG_ + VN_*C_ + 255)/256, 256, 0, stream>>>(
      hyper_joint, conv_d_w, out + (size_t)N_*O_*T_*V_, wf16);
  hyper_b1_kernel<<<N_*K_, 64, 0, stream>>>(pooled, to_v_w, to_v_b,
                                            to_w1_w, to_w1_b, incid, w1);
  hyper_b2_kernel<<<N_*K_, 32, 0, stream>>>(incid, w1, to_w2_w, to_w2_b,
                                            adjacency, edge_importance,
                                            alpha, adjf);
  fused_gemm_kernel<<<N_*K_*(T_/TTILE), 256, 0, stream>>>(
      x, hyper_joint, wf16, conv_d_b, bn_gamma, bn_beta, adjf, out);
}
